// DenseDilatedKnnGraph_81638738362638
// MI455X (gfx1250) — matrix operand order repaired and checked by reading, not compiled, not run on hardware
//
#include <hip/hip_runtime.h>
#include <hip/hip_fp16.h>

typedef __attribute__((ext_vector_type(16))) _Float16 v16h;
typedef __attribute__((ext_vector_type(8)))  float    v8f;
typedef __attribute__((ext_vector_type(4)))  int      v4i;

// Problem constants (from reference setup_inputs)
#define BB    4
#define CC    128
#define NPTS  8192
#define KK    16
#define TILES (NPTS / 16)        // 512 column tiles per batch

// Staged B-tile layout in LDS: 16 rows x 256B, padded to 272B/row so that
// consecutive rows start on different bank quads (272B = 68 dwords, 68%64=4).
#define ROW_STRIDE 272
#define BUF_SZ     (16 * ROW_STRIDE)   // 4352 B per buffer, 3 buffers

#if defined(__has_builtin)
#if __has_builtin(__builtin_amdgcn_global_load_async_to_lds_b128)
#define HAVE_ASYNC_LDS 1
#endif
#endif

// Address-space-qualified pointer types for the async builtin
// (signature revealed by the compiler: non-const v4i global ptr, v4i LDS ptr).
typedef __attribute__((address_space(1))) v4i* gptr_v4i;
typedef __attribute__((address_space(3))) v4i* lptr_v4i;

union Frag {
    v16h h;
    uint4 q[2];
};

static __device__ __forceinline__ void wait_asynccnt_le1() {
#if defined(HAVE_ASYNC_LDS)
#if __has_builtin(__builtin_amdgcn_s_wait_asynccnt)
    __builtin_amdgcn_s_wait_asynccnt(1);
#else
    asm volatile("s_wait_asynccnt 0x1" ::: "memory");
#endif
#endif
}

// ---------------------------------------------------------------------------
// Kernel 1: L2-normalize along channel dim, pack to f16 point-major (b,n,c),
// and record sq[b][n] = sum(xt^2) (== 1 up to rounding, kept for exactness).
// ---------------------------------------------------------------------------
__global__ __launch_bounds__(256)
void knn_normalize_pack(const float* __restrict__ x,
                        _Float16* __restrict__ xt,
                        float* __restrict__ sq) {
    int t = blockIdx.x * blockDim.x + threadIdx.x;   // 0 .. B*N-1
    if (t >= BB * NPTS) return;
    int b = t / NPTS;
    int n = t - b * NPTS;

    const float* xb = x + (size_t)b * CC * NPTS + n; // x[b][c][n], stride N over c
    float s = 0.0f;
#pragma unroll 8
    for (int c = 0; c < CC; ++c) {
        float v = xb[(size_t)c * NPTS];
        s += v * v;
    }
    float norm = sqrtf(s);
    float inv  = 1.0f / fmaxf(norm, 1e-12f);
    sq[t] = s * inv * inv;

    _Float16* dst = xt + (size_t)t * CC;
#pragma unroll
    for (int c0 = 0; c0 < CC; c0 += 8) {
        union { _Float16 h[8]; uint4 q; } u;
#pragma unroll
        for (int j = 0; j < 8; ++j)
            u.h[j] = (_Float16)(xb[(size_t)(c0 + j) * NPTS] * inv);
        *(uint4*)(dst + c0) = u.q;
    }
}

// ---------------------------------------------------------------------------
// Kernel 2: fused WMMA inner-product sweep + streaming top-16 per row.
// One wave (wave32) per 16-row tile; 8 waves per block, all sharing batch b
// and the column-tile schedule, so B tiles are staged into LDS once per block
// via async-to-LDS copies (triple-buffered, 2 tiles ahead).
//
// A frag (16x32 f16): lane = row (m = lane&15), h = lane>>4:
//   VGPR j: K = h*16 + 2j        -> two 16B at c = kk*32 + h*16 (+8)
// B frag (32x16 f16): lane = col, h = lane>>4:
//   VGPR j: K = h*16 + 2j        -> two 16B at c = kk*32 + h*16 (+8)
// C/D (16x16 f32): VGPR v, lane l -> row M = v + 8*(l>=16), col N = l&15.
// ---------------------------------------------------------------------------
__global__ __launch_bounds__(256)
void knn_wmma_topk(const _Float16* __restrict__ xt,
                   const float* __restrict__ sq,
                   int* __restrict__ out) {
#if defined(HAVE_ASYNC_LDS)
    __shared__ char  ldsB[3 * BUF_SZ];       // staged B tiles (triple buffer)
#endif
    __shared__ float ldsTile[8 * 16 * 17];   // per-wave 16x16 dist tile, padded
    __shared__ float ldsD[8][32][KK];        // per-wave per-lane sorted dists
    __shared__ int   ldsI[8][32][KK];        // per-wave per-lane sorted indices

    const int lane = threadIdx.x & 31;
    const int wv   = threadIdx.x >> 5;
    const int gw   = blockIdx.x * 8 + wv;    // global wave id
    const int b    = gw / TILES;             // block-uniform (64 blocks/batch)
    const int tile = gw - b * TILES;
    const int n0   = tile * 16;
    const int r    = lane & 15;
    const int h    = lane >> 4;

    const _Float16* xb  = xt + (size_t)b * NPTS * CC;
    const float*    sqb = sq + (size_t)b * NPTS;

#if defined(HAVE_ASYNC_LDS)
    // Cooperative async stage of one 16x256B column tile into LDS buffer.
    const int srow = threadIdx.x >> 4;       // 0..15
    const int sjj  = threadIdx.x & 15;       // 0..15 (16B chunk within row)
    auto stage = [&](int mtile, int bufidx) {
        const char* src = (const char*)xb +
                          (size_t)(mtile * 16 + srow) * (CC * 2) + sjj * 16;
        char* dst = &ldsB[bufidx * BUF_SZ + srow * ROW_STRIDE + sjj * 16];
        __builtin_amdgcn_global_load_async_to_lds_b128(
            (gptr_v4i)(uintptr_t)src, (lptr_v4i)(uintptr_t)dst, 0, 0);
    };
    stage(0, 0);
    stage(1, 1);
#endif

    // --- A fragments: rows n0..n0+15, all of C=128 (4 K-chunks of 32) ---
    Frag a[4];
    {
        const _Float16* arow = xb + (size_t)(n0 + r) * CC;
#pragma unroll
        for (int kk = 0; kk < 4; ++kk) {
            int c = kk * 32 + h * 16;
            a[kk].q[0] = *(const uint4*)(arow + c);
            a[kk].q[1] = *(const uint4*)(arow + c + 8);
        }
    }

    // --- per-lane top-16 list (sorted ascending; ties keep earlier/lower idx) ---
    float dd[KK];
    int   ii[KK];
#pragma unroll
    for (int i = 0; i < KK; ++i) { dd[i] = 3.0e38f; ii[i] = 0; }

    float* myTile = ldsTile + wv * (16 * 17);

    for (int mt = 0; mt < TILES; ++mt) {
        const int m0 = mt * 16;

#if defined(HAVE_ASYNC_LDS)
        // Current buffer's copy is complete (only the newest may be in flight).
        wait_asynccnt_le1();
        __syncthreads();
        // Prefetch 2 tiles ahead; wrap the tail to tile 0 to keep ASYNCcnt
        // bookkeeping uniform (those buffers are never read).
        {
            int pf = (mt + 2 < TILES) ? (mt + 2) : 0;
            stage(pf, (mt + 2) % 3);
        }
        // --- B fragments from staged LDS tile ---
        Frag bf[4];
        {
            const char* bbase = ldsB + (mt % 3) * BUF_SZ + r * ROW_STRIDE + h * 32;
#pragma unroll
            for (int kk = 0; kk < 4; ++kk) {
                bf[kk].q[0] = *(const uint4*)(bbase + kk * 64);
                bf[kk].q[1] = *(const uint4*)(bbase + kk * 64 + 16);
            }
        }
#else
        // Fallback: per-wave direct global loads (L2-resident).
        Frag bf[4];
        {
            const _Float16* brow = xb + (size_t)(m0 + r) * CC;
#pragma unroll
            for (int kk = 0; kk < 4; ++kk) {
                int c = kk * 32 + h * 16;
                bf[kk].q[0] = *(const uint4*)(brow + c);
                bf[kk].q[1] = *(const uint4*)(brow + c + 8);
            }
        }
#endif

        // --- 16x16 inner products over K=128 via 4 WMMAs ---
        v8f acc = {};
#pragma unroll
        for (int kk = 0; kk < 4; ++kk)
            acc = __builtin_amdgcn_wmma_f32_16x16x32_f16(
                false, a[kk].h, false, bf[kk].h, (short)0, acc, false, false);

        // --- epilogue: d = sq[m] - 2*inner (sq[n] is row-constant, drop it) ---
        float sqc = sqb[m0 + r];   // this lane's column
#pragma unroll
        for (int v = 0; v < 8; ++v) {
            int row = v + 8 * h;
            myTile[row * 17 + r] = fmaf(-2.0f, acc[v], sqc);
        }
        asm volatile("s_wait_dscnt 0" ::: "memory");

        // --- streaming selection: lane handles row r, columns h*8 .. h*8+7 ---
#pragma unroll
        for (int j = 0; j < 8; ++j) {
            int   col = h * 8 + j;
            float d   = myTile[r * 17 + col];
            int   idx = m0 + col;
            if (d < dd[KK - 1]) {
                dd[KK - 1] = d; ii[KK - 1] = idx;
#pragma unroll
                for (int i = KK - 1; i > 0; --i) {
                    if (dd[i] < dd[i - 1]) {
                        float td = dd[i]; dd[i] = dd[i - 1]; dd[i - 1] = td;
                        int   ti = ii[i]; ii[i] = ii[i - 1]; ii[i - 1] = ti;
                    }
                }
            }
        }
        asm volatile("s_wait_dscnt 0" ::: "memory");
    }

    // --- publish per-lane sorted lists, merge the two half-lane lists per row ---
#pragma unroll
    for (int i = 0; i < KK; ++i) {
        ldsD[wv][lane][i] = dd[i];
        ldsI[wv][lane][i] = ii[i];
    }
    asm volatile("s_wait_dscnt 0" ::: "memory");

    if (h == 0) {
        int ngl = n0 + r;
        int* nnOut = out + ((size_t)b * NPTS + ngl) * KK;
        int* ctOut = nnOut + (size_t)BB * NPTS * KK;
        int ia = 0, ib = 0;
#pragma unroll
        for (int t = 0; t < KK; ++t) {
            float da  = ldsD[wv][r][ia];
            float db2 = ldsD[wv][r + 16][ib];
            int   iaI = ldsI[wv][r][ia];
            int   ibI = ldsI[wv][r + 16][ib];
            bool takeA = (da < db2) || ((da == db2) && (iaI < ibI));
            nnOut[t] = takeA ? iaI : ibI;
            ctOut[t] = ngl;
            if (takeA) ++ia; else ++ib;
        }
    }
}

// ---------------------------------------------------------------------------
extern "C" void kernel_launch(void* const* d_in, const int* in_sizes, int n_in,
                              void* d_out, int out_size, void* d_ws, size_t ws_size,
                              hipStream_t stream) {
    const float* x = (const float*)d_in[0];
    int* out = (int*)d_out;

    // workspace layout: sq (B*N f32) | xt (B*N*C f16)
    float*    sq = (float*)d_ws;
    _Float16* xt = (_Float16*)((char*)d_ws + (size_t)BB * NPTS * sizeof(float));

    {
        int threads = BB * NPTS;
        int blk = 256;
        knn_normalize_pack<<<(threads + blk - 1) / blk, blk, 0, stream>>>(x, xt, sq);
    }
    {
        int waves  = BB * TILES;         // 2048 waves
        int blocks = waves / 8;          // 8 waves / 256-thread block
        knn_wmma_topk<<<blocks, 256, 0, stream>>>(xt, sq, out);
    }
}
